// AGCRN_13649406066690
// MI455X (gfx1250) — compile-verified, matrix operand order checked by reading, not run on hardware
//
#include <hip/hip_runtime.h>
#include <math.h>

// ---------------------------------------------------------------------------
// AGCRN on gfx1250: bf16 WMMA (v_wmma_f32_16x16x32_bf16) for all GEMMs.
// B=16, N=1024, T=24, C_IN=1, EMBED=32, CHEB=3, HID=64, OUT=36
// ---------------------------------------------------------------------------

typedef __attribute__((ext_vector_type(16))) __bf16 v16bf;
typedef __attribute__((ext_vector_type(8)))  float  v8f;

__device__ inline v8f wmma_bf16(v16bf a, v16bf b, v8f c) {
  return __builtin_amdgcn_wmma_f32_16x16x32_bf16(
      /*neg_a=*/false, a, /*neg_b=*/false, b,
      /*c_mod=*/(short)0, c, /*reuse_a=*/false, /*reuse_b=*/false);
}

__device__ inline float sigf(float x) { return 1.0f / (1.0f + __expf(-x)); }

// ------------------------- Precompute kernels ------------------------------

// G[i][j] = sum_d E[i,d]*E[j,d]  (1024x1024)
__global__ void k_gram(const float* __restrict__ E, float* __restrict__ G) {
  const int total = 1024 * 1024;
  for (int idx = blockIdx.x * blockDim.x + threadIdx.x; idx < total;
       idx += gridDim.x * blockDim.x) {
    int j = idx & 1023, i = idx >> 10;
    float acc = 0.f;
    const float* a = E + i * 32;
    const float* b = E + j * 32;
#pragma unroll
    for (int d = 0; d < 32; ++d) acc += a[d] * b[d];
    G[idx] = acc;
  }
}

// S1[row] = softmax(relu(G[row]))  -> bf16
__global__ __launch_bounds__(256) void k_rowsoftmax(const float* __restrict__ G,
                                                    __bf16* __restrict__ S) {
  __shared__ float red[256];
  const int row = blockIdx.x, tid = threadIdx.x;
  const float* g = G + (size_t)row * 1024;
  float v[4];
  float mx = 0.f;
#pragma unroll
  for (int q = 0; q < 4; ++q) {
    float x = fmaxf(g[tid + q * 256], 0.f);
    v[q] = x;
    mx = fmaxf(mx, x);
  }
  red[tid] = mx;
  __syncthreads();
  for (int s = 128; s > 0; s >>= 1) {
    if (tid < s) red[tid] = fmaxf(red[tid], red[tid + s]);
    __syncthreads();
  }
  mx = red[0];
  __syncthreads();
  float sum = 0.f;
#pragma unroll
  for (int q = 0; q < 4; ++q) {
    float e = __expf(v[q] - mx);
    v[q] = e;
    sum += e;
  }
  red[tid] = sum;
  __syncthreads();
  for (int s = 128; s > 0; s >>= 1) {
    if (tid < s) red[tid] += red[tid + s];
    __syncthreads();
  }
  float inv = 1.f / red[0];
  __bf16* o = S + (size_t)row * 1024;
#pragma unroll
  for (int q = 0; q < 4; ++q) o[tid + q * 256] = (__bf16)(v[q] * inv);
}

// S2 = 2*A*A - I  (bf16 WMMA, f32 accumulate).  128x64 tile, 32-k LDS staging.
__global__ __launch_bounds__(256) void k_gemm_AA(const __bf16* __restrict__ A,
                                                 __bf16* __restrict__ S2) {
  const int m0 = blockIdx.y * 128;
  const int j0 = blockIdx.x * 64;
  const int tid = threadIdx.x;
  const int wave = tid >> 5, lane = tid & 31;
  const int half = lane >> 4, lm = lane & 15;
  const int wm = (wave & 3) * 32;
  const int wj = (wave >> 2) * 32;

  __shared__ __bf16 lA[128 * 32];
  __shared__ __bf16 lB[64 * 32];

  v8f acc[2][2] = {};
  for (int kb = 0; kb < 1024; kb += 32) {
    {
      int row = tid >> 1, c = (tid & 1) * 16;
      const uint4* s = (const uint4*)(A + (size_t)(m0 + row) * 1024 + kb + c);
      uint4* d = (uint4*)(lA + row * 32 + c);
      d[0] = s[0];
      d[1] = s[1];
    }
    {
      int kk = tid >> 3, jo = (tid & 7) * 8;
      uint4 v = *(const uint4*)(A + (size_t)(kb + kk) * 1024 + j0 + jo);
      const __bf16* pv = (const __bf16*)&v;
#pragma unroll
      for (int q = 0; q < 8; ++q) lB[(jo + q) * 32 + kk] = pv[q];
    }
    __syncthreads();
    v16bf a0 = *(const v16bf*)(lA + (wm + lm) * 32 + half * 16);
    v16bf a1 = *(const v16bf*)(lA + (wm + 16 + lm) * 32 + half * 16);
    v16bf b0 = *(const v16bf*)(lB + (wj + lm) * 32 + half * 16);
    v16bf b1 = *(const v16bf*)(lB + (wj + 16 + lm) * 32 + half * 16);
    acc[0][0] = wmma_bf16(a0, b0, acc[0][0]);
    acc[0][1] = wmma_bf16(a0, b1, acc[0][1]);
    acc[1][0] = wmma_bf16(a1, b0, acc[1][0]);
    acc[1][1] = wmma_bf16(a1, b1, acc[1][1]);
    __syncthreads();
  }
#pragma unroll
  for (int ti = 0; ti < 2; ++ti)
#pragma unroll
    for (int tj = 0; tj < 2; ++tj) {
      int j = j0 + wj + tj * 16 + lm;
#pragma unroll
      for (int r = 0; r < 8; ++r) {
        int m = m0 + wm + ti * 16 + r + 8 * half;
        float dval = 2.0f * acc[ti][tj][r] - ((m == j) ? 1.0f : 0.0f);
        S2[(size_t)m * 1024 + j] = (__bf16)dval;
      }
    }
}

// Per-node weight expansion: Wt[n][o][kk] = sum_d E[n,d]*W[d,k,i,o], kk=k*I+i,
// zero-padded to Ktot.  Stored bf16.
__global__ void k_wprep(const float* __restrict__ E, const float* __restrict__ W,
                        __bf16* __restrict__ Wt, int I, int O, int Ktot) {
  const size_t total = (size_t)1024 * O * Ktot;
  const size_t dstride = (size_t)3 * I * O;
  for (size_t idx = (size_t)blockIdx.x * blockDim.x + threadIdx.x; idx < total;
       idx += (size_t)gridDim.x * blockDim.x) {
    int kk = (int)(idx % Ktot);
    size_t t2 = idx / Ktot;
    int o = (int)(t2 % O);
    int n = (int)(t2 / O);
    float acc = 0.f;
    if (kk < 3 * I) {
      int k = kk / I, i = kk % I;
      const float* wp = W + ((size_t)k * I + i) * O + o;
      const float* ep = E + n * 32;
#pragma unroll
      for (int d = 0; d < 32; ++d) acc += ep[d] * wp[d * dstride];
    }
    Wt[idx] = (__bf16)acc;
  }
}

// Bias expansion: Bd[n][o] = sum_d E[n,d]*Bs[d,o]
__global__ void k_bprep(const float* __restrict__ E, const float* __restrict__ Bs,
                        float* __restrict__ Bd, int O) {
  const int total = 1024 * O;
  for (int idx = blockIdx.x * blockDim.x + threadIdx.x; idx < total;
       idx += gridDim.x * blockDim.x) {
    int o = idx % O, n = idx / O;
    float acc = 0.f;
    const float* ep = E + n * 32;
#pragma unroll
    for (int d = 0; d < 32; ++d) acc += ep[d] * Bs[d * O + o];
    Bd[idx] = acc;
  }
}

// ------------------------- Per-step kernels --------------------------------

// Layer-0 gate input: X = concat([x_t, h0]) -> Xb (GEMM B matrix) + XG slot0
__global__ void k_prep0(const float* __restrict__ x, const float* __restrict__ h0,
                        __bf16* __restrict__ Xb, __bf16* XG, int t) {
  const int total = 1024 * 16 * 65;
  for (int idx = blockIdx.x * blockDim.x + threadIdx.x; idx < total;
       idx += gridDim.x * blockDim.x) {
    int i = idx % 65;
    int r = idx / 65;
    int b = r % 16, n = r / 16;
    float val = (i == 0) ? x[((size_t)(b * 1024 + n)) * 24 + t]
                         : h0[((size_t)n * 16 + b) * 64 + (i - 1)];
    __bf16 v = (__bf16)val;
    Xb[(size_t)n * 1088 + b * 65 + i] = v;
    XG[((size_t)n * 16 + b) * 224 + i] = v;
  }
}

// Layer-1 gate input: X = concat([h0_new, h1])
__global__ void k_prep1(const float* __restrict__ h0, const float* __restrict__ h1,
                        __bf16* __restrict__ Xb, __bf16* XG) {
  const int total = 1024 * 16 * 128;
  for (int idx = blockIdx.x * blockDim.x + threadIdx.x; idx < total;
       idx += gridDim.x * blockDim.x) {
    int i = idx % 128;
    int r = idx / 128;
    int b = r % 16, n = r / 16;
    float val = (i < 64) ? h0[((size_t)n * 16 + b) * 64 + i]
                         : h1[((size_t)n * 16 + b) * 64 + (i - 64)];
    __bf16 v = (__bf16)val;
    Xb[(size_t)n * 2048 + b * 128 + i] = v;
    XG[((size_t)n * 16 + b) * 384 + i] = v;
  }
}

// Graph conv: XG[n][b][slot_k*I + i] = sum_m S_k[n,m] * Xb[m][b*I+i], k=1,2.
// Tile 128(n) x 64(j), j = b*I+i; bf16 WMMA, f32 accumulate, bf16 scatter-out.
__global__ __launch_bounds__(256) void k_gconv(const __bf16* __restrict__ S1,
                                               const __bf16* __restrict__ S2,
                                               const __bf16* __restrict__ Xb,
                                               __bf16* __restrict__ XG, int Jpad,
                                               int Inode, int Ktot) {
  const __bf16* S = (blockIdx.z == 0) ? S1 : S2;
  const int slotoff = (1 + blockIdx.z) * Inode;
  const int m0 = blockIdx.y * 128;
  const int j0 = blockIdx.x * 64;
  const int tid = threadIdx.x;
  const int wave = tid >> 5, lane = tid & 31;
  const int half = lane >> 4, lm = lane & 15;
  const int wm = (wave & 3) * 32;
  const int wj = (wave >> 2) * 32;

  __shared__ __bf16 lA[128 * 32];
  __shared__ __bf16 lB[64 * 32];

  v8f acc[2][2] = {};
  for (int kb = 0; kb < 1024; kb += 32) {
    {
      int row = tid >> 1, c = (tid & 1) * 16;
      const uint4* s = (const uint4*)(S + (size_t)(m0 + row) * 1024 + kb + c);
      uint4* d = (uint4*)(lA + row * 32 + c);
      d[0] = s[0];
      d[1] = s[1];
    }
    {
      int kk = tid >> 3, jo = (tid & 7) * 8;
      uint4 v = *(const uint4*)(Xb + (size_t)(kb + kk) * Jpad + j0 + jo);
      const __bf16* pv = (const __bf16*)&v;
#pragma unroll
      for (int q = 0; q < 8; ++q) lB[(jo + q) * 32 + kk] = pv[q];
    }
    __syncthreads();
    v16bf a0 = *(const v16bf*)(lA + (wm + lm) * 32 + half * 16);
    v16bf a1 = *(const v16bf*)(lA + (wm + 16 + lm) * 32 + half * 16);
    v16bf b0 = *(const v16bf*)(lB + (wj + lm) * 32 + half * 16);
    v16bf b1 = *(const v16bf*)(lB + (wj + 16 + lm) * 32 + half * 16);
    acc[0][0] = wmma_bf16(a0, b0, acc[0][0]);
    acc[0][1] = wmma_bf16(a0, b1, acc[0][1]);
    acc[1][0] = wmma_bf16(a1, b0, acc[1][0]);
    acc[1][1] = wmma_bf16(a1, b1, acc[1][1]);
    __syncthreads();
  }
#pragma unroll
  for (int ti = 0; ti < 2; ++ti)
#pragma unroll
    for (int tj = 0; tj < 2; ++tj) {
      int j = j0 + wj + tj * 16 + lm;
      int b = j / Inode;
      int i = j - b * Inode;
      if (b < 16) {
#pragma unroll
        for (int r = 0; r < 8; ++r) {
          int m = m0 + wm + ti * 16 + r + 8 * half;
          XG[((size_t)m * 16 + b) * Ktot + slotoff + i] = (__bf16)acc[ti][tj][r];
        }
      }
    }
}

// Gate: zr = sigmoid(XA(16xKtot) x Wt_n(Ktot x 128) + bg_n); z -> z*h into
// Xb & XG slot0 (building update input), r -> Rbuf.  One block per node.
__global__ __launch_bounds__(256) void k_gate(const __bf16* XG,
                                              const __bf16* __restrict__ Wt,
                                              const float* __restrict__ bg,
                                              const float* __restrict__ h,
                                              float* __restrict__ R, __bf16* Xb,
                                              int Ktot, int Jpad, int Inode,
                                              int xtoff) {
  const int n = blockIdx.x;
  const int wave = threadIdx.x >> 5, lane = threadIdx.x & 31;
  const int half = lane >> 4, lm = lane & 15;
  const int o0 = wave * 16;
  const __bf16* Arow = XG + ((size_t)n * 16 + lm) * Ktot + half * 16;
  const __bf16* Brow = Wt + ((size_t)n * 128 + o0 + lm) * Ktot + half * 16;
  v8f acc = {};
  for (int ko = 0; ko < Ktot; ko += 32) {
    v16bf a = *(const v16bf*)(Arow + ko);
    v16bf b = *(const v16bf*)(Brow + ko);
    acc = wmma_bf16(a, b, acc);
  }
  __syncthreads();  // all waves done reading XG[n] before slot0 is overwritten
  const int o = o0 + lm;
  const float bias = bg[n * 128 + o];
#pragma unroll
  for (int r = 0; r < 8; ++r) {
    int bidx = r + 8 * half;
    float s = sigf(acc[r] + bias);
    size_t hrow = ((size_t)n * 16 + bidx) * 64;
    if (o < 64) {
      float zh = s * h[hrow + o];
      __bf16 v = (__bf16)zh;
      Xb[(size_t)n * Jpad + bidx * Inode + xtoff + o] = v;
      ((__bf16*)XG)[((size_t)n * 16 + bidx) * Ktot + xtoff + o] = v;
    } else {
      R[hrow + (o - 64)] = s;
    }
  }
}

// Update: hc = tanh(XA x Wu_n + bu_n); h = r*h + (1-r)*hc. One block per node.
__global__ __launch_bounds__(128) void k_update(const __bf16* __restrict__ XG,
                                                const __bf16* __restrict__ Wt,
                                                const float* __restrict__ bu,
                                                const float* __restrict__ R,
                                                float* __restrict__ h, int Ktot) {
  const int n = blockIdx.x;
  const int wave = threadIdx.x >> 5, lane = threadIdx.x & 31;
  const int half = lane >> 4, lm = lane & 15;
  const int o0 = wave * 16;
  const __bf16* Arow = XG + ((size_t)n * 16 + lm) * Ktot + half * 16;
  const __bf16* Brow = Wt + ((size_t)n * 64 + o0 + lm) * Ktot + half * 16;
  v8f acc = {};
  for (int ko = 0; ko < Ktot; ko += 32) {
    v16bf a = *(const v16bf*)(Arow + ko);
    v16bf b = *(const v16bf*)(Brow + ko);
    acc = wmma_bf16(a, b, acc);
  }
  const int o = o0 + lm;
  const float bias = bu[n * 64 + o];
#pragma unroll
  for (int r = 0; r < 8; ++r) {
    int bidx = r + 8 * half;
    float hc = tanhf(acc[r] + bias);
    size_t idx = ((size_t)n * 16 + bidx) * 64 + o;
    float rr = R[idx];
    h[idx] = rr * h[idx] + (1.0f - rr) * hc;
  }
}

// Final projection: out[b,n,o] = h1[n,b,:] . conv_w[o,:] + conv_b[o]
__global__ void k_final(const float* __restrict__ h1, const float* __restrict__ cw,
                        const float* __restrict__ cb, float* __restrict__ out) {
  const int total = 16 * 1024 * 36;
  for (int idx = blockIdx.x * blockDim.x + threadIdx.x; idx < total;
       idx += gridDim.x * blockDim.x) {
    int o = idx % 36;
    int r = idx / 36;
    int n = r % 1024, b = r / 1024;
    const float* hp = h1 + ((size_t)n * 16 + b) * 64;
    const float* wp = cw + o * 64;
    float acc = cb[o];
#pragma unroll
    for (int c = 0; c < 64; ++c) acc += hp[c] * wp[c];
    out[idx] = acc;
  }
}

// ---------------------------------------------------------------------------

extern "C" void kernel_launch(void* const* d_in, const int* in_sizes, int n_in,
                              void* d_out, int out_size, void* d_ws,
                              size_t ws_size, hipStream_t stream) {
  (void)in_sizes; (void)n_in; (void)out_size;
  const float* x   = (const float*)d_in[0];
  const float* E   = (const float*)d_in[1];
  const float* w0g = (const float*)d_in[2];
  const float* b0g = (const float*)d_in[3];
  const float* w0u = (const float*)d_in[4];
  const float* b0u = (const float*)d_in[5];
  const float* w1g = (const float*)d_in[6];
  const float* b1g = (const float*)d_in[7];
  const float* w1u = (const float*)d_in[8];
  const float* b1u = (const float*)d_in[9];
  const float* cw  = (const float*)d_in[10];
  const float* cb  = (const float*)d_in[11];
  float* out = (float*)d_out;

  char* ws = (char*)d_ws;
  size_t off = 0;
  auto take = [&](size_t bytes) -> char* {
    char* p = ws + off;
    off += (bytes + 255) & ~(size_t)255;
    return p;
  };
  float*  G    = (float*)take(1024ull * 1024 * 4);
  __bf16* S1   = (__bf16*)take(1024ull * 1024 * 2);
  __bf16* S2   = (__bf16*)take(1024ull * 1024 * 2);
  __bf16* Wt0g = (__bf16*)take(1024ull * 128 * 224 * 2);
  __bf16* Wt0u = (__bf16*)take(1024ull * 64 * 224 * 2);
  __bf16* Wt1g = (__bf16*)take(1024ull * 128 * 384 * 2);
  __bf16* Wt1u = (__bf16*)take(1024ull * 64 * 384 * 2);
  float*  bg0  = (float*)take(1024ull * 128 * 4);
  float*  bu0  = (float*)take(1024ull * 64 * 4);
  float*  bg1  = (float*)take(1024ull * 128 * 4);
  float*  bu1  = (float*)take(1024ull * 64 * 4);
  float*  h0   = (float*)take(1024ull * 16 * 64 * 4);
  float*  h1   = (float*)take(1024ull * 16 * 64 * 4);
  float*  Rb   = (float*)take(1024ull * 16 * 64 * 4);
  __bf16* Xb0  = (__bf16*)take(1024ull * 1088 * 2);
  __bf16* Xb1  = (__bf16*)take(1024ull * 2048 * 2);
  __bf16* XG0  = (__bf16*)take(1024ull * 16 * 224 * 2);
  __bf16* XG1  = (__bf16*)take(1024ull * 16 * 384 * 2);
  if (off > ws_size) return;  // insufficient scratch

  // Deterministic per-launch init (h starts at zero; XG pads must be finite).
  hipMemsetAsync(h0, 0, 1024ull * 16 * 64 * 4, stream);
  hipMemsetAsync(h1, 0, 1024ull * 16 * 64 * 4, stream);
  hipMemsetAsync(XG0, 0, 1024ull * 16 * 224 * 2, stream);
  hipMemsetAsync(XG1, 0, 1024ull * 16 * 384 * 2, stream);
  hipMemsetAsync(Xb0, 0, 1024ull * 1088 * 2, stream);
  hipMemsetAsync(Xb1, 0, 1024ull * 2048 * 2, stream);

  // --- Precompute supports and per-node weights/biases ---
  k_gram<<<4096, 256, 0, stream>>>(E, G);
  k_rowsoftmax<<<1024, 256, 0, stream>>>(G, S1);
  k_gemm_AA<<<dim3(16, 8), 256, 0, stream>>>(S1, S2);
  k_wprep<<<16384, 256, 0, stream>>>(E, w0g, Wt0g, 65, 128, 224);
  k_wprep<<<16384, 256, 0, stream>>>(E, w0u, Wt0u, 65, 64, 224);
  k_wprep<<<16384, 256, 0, stream>>>(E, w1g, Wt1g, 128, 128, 384);
  k_wprep<<<16384, 256, 0, stream>>>(E, w1u, Wt1u, 128, 64, 384);
  k_bprep<<<512, 256, 0, stream>>>(E, b0g, bg0, 128);
  k_bprep<<<256, 256, 0, stream>>>(E, b0u, bu0, 64);
  k_bprep<<<512, 256, 0, stream>>>(E, b1g, bg1, 128);
  k_bprep<<<256, 256, 0, stream>>>(E, b1u, bu1, 64);

  // --- 24 GRU steps, two stacked layers ---
  for (int t = 0; t < 24; ++t) {
    // layer 0 (I=65, Ktot=224, Jpad=1088, xtoff=1)
    k_prep0<<<4160, 256, 0, stream>>>(x, h0, Xb0, XG0, t);
    k_gconv<<<dim3(17, 8, 2), 256, 0, stream>>>(S1, S2, Xb0, XG0, 1088, 65, 224);
    k_gate<<<1024, 256, 0, stream>>>(XG0, Wt0g, bg0, h0, Rb, Xb0, 224, 1088, 65, 1);
    k_gconv<<<dim3(17, 8, 2), 256, 0, stream>>>(S1, S2, Xb0, XG0, 1088, 65, 224);
    k_update<<<1024, 128, 0, stream>>>(XG0, Wt0u, bu0, Rb, h0, 224);
    // layer 1 (I=128, Ktot=384, Jpad=2048, xtoff=64)
    k_prep1<<<8192, 256, 0, stream>>>(h0, h1, Xb1, XG1);
    k_gconv<<<dim3(32, 8, 2), 256, 0, stream>>>(S1, S2, Xb1, XG1, 2048, 128, 384);
    k_gate<<<1024, 256, 0, stream>>>(XG1, Wt1g, bg1, h1, Rb, Xb1, 384, 2048, 128, 64);
    k_gconv<<<dim3(32, 8, 2), 256, 0, stream>>>(S1, S2, Xb1, XG1, 2048, 128, 384);
    k_update<<<1024, 128, 0, stream>>>(XG1, Wt1u, bu1, Rb, h1, 384);
  }

  k_final<<<2304, 256, 0, stream>>>(h1, cw, cb, out);
}
